// Net_2353642078708
// MI455X (gfx1250) — matrix operand order repaired and checked by reading, not compiled, not run on hardware
//
#include <hip/hip_runtime.h>
#include <hip/hip_bf16.h>
#include <stdint.h>

// ---------------------------------------------------------------------------
// MI455X (gfx1250) implementation. wave32, WMMA bf16 16x16x32, f32 accum.
// Compile-time K => fully unrolled, register-renamed ping-pong fragment
// buffers (no v_mov rotation), immediate-offset loads. Activations use
// v_rcp_f32/v_exp_f32 (TRANS ops co-issue with the matrix pipe), not the
// IEEE div slow path.
// ---------------------------------------------------------------------------

typedef __attribute__((ext_vector_type(8)))  __bf16 v8bf;
typedef __attribute__((ext_vector_type(16))) __bf16 v16bf;
typedef __attribute__((ext_vector_type(8)))  float  v8f;

#define BDIM 16384
#define TSTEPS 96
#define HDIM 256

__device__ __forceinline__ v16bf ld_bf16_contig16(const __bf16* p) {
  v8bf lo = *reinterpret_cast<const v8bf*>(p);
  v8bf hi = *reinterpret_cast<const v8bf*>(p + 8);
  return __builtin_shufflevector(lo, hi, 0,1,2,3,4,5,6,7,8,9,10,11,12,13,14,15);
}
__device__ __forceinline__ v16bf ld_bf16_8x2(const __bf16* p0, const __bf16* p1) {
  v8bf lo = *reinterpret_cast<const v8bf*>(p0);
  v8bf hi = *reinterpret_cast<const v8bf*>(p1);
  return __builtin_shufflevector(lo, hi, 0,1,2,3,4,5,6,7,8,9,10,11,12,13,14,15);
}
__device__ __forceinline__ v8f wmma_bf16(v16bf a, v16bf b, v8f c) {
  return __builtin_amdgcn_wmma_f32_16x16x32_bf16(false, a, false, b, (short)0, c,
                                                 false, false);
}
// Fast activations: v_exp_f32 + v_rcp_f32 (TRANS pipe), no IEEE-div fixup.
__device__ __forceinline__ float sigm(float x) {
  return __builtin_amdgcn_rcpf(1.f + __expf(-x));
}
__device__ __forceinline__ float fast_tanh(float x) {
  return 2.f * __builtin_amdgcn_rcpf(1.f + __expf(-2.f * x)) - 1.f;
}

// ---------------------------------------------------------------------------
// Weight packing: fp32 [N][K] row-major -> bf16 [N][Kp], zero-padded K.
// ---------------------------------------------------------------------------
__global__ void pack_weight(const float* __restrict__ W, __bf16* __restrict__ Wp,
                            int N, int K, int Kp) {
  int i = blockIdx.x * 256 + threadIdx.x;
  if (i >= N * Kp) return;
  int n = i / Kp, k = i - n * Kp;
  Wp[i] = (__bf16)(k < K ? W[n * K + k] : 0.f);
}

// Pack [w_ih(1024x56) | pad8 | w_hh(1024x256)] -> Wcat bf16 [1024][320];
// combined bias bcat = b_ih + b_hh.
__global__ void pack_lstm(const float* __restrict__ w_ih, const float* __restrict__ w_hh,
                          const float* __restrict__ b_ih, const float* __restrict__ b_hh,
                          __bf16* __restrict__ Wcat, float* __restrict__ bcat) {
  int i = blockIdx.x * 256 + threadIdx.x;
  if (i < 1024) bcat[i] = b_ih[i] + b_hh[i];
  if (i >= 1024 * 320) return;
  int n = i / 320, k = i - n * 320;
  float v = 0.f;
  if (k < 56) v = w_ih[n * 56 + k];
  else if (k >= 64) v = w_hh[n * 256 + (k - 64)];
  Wcat[i] = (__bf16)v;
}

// Per-row spatial prep: pack seg-GEMM input A_sp bf16 [B][1280] and fill the
// X1 concat buffer [B][576]: cols 0..1 = sl, 2..9 = st, 522..526 = rest,
// 527..575 = 0. (cols 10..521 written later by the seg GEMM.)
__global__ void pack_spatial(const float* __restrict__ sp,
                             const float* __restrict__ sl_w, const float* __restrict__ sl_b,
                             const float* __restrict__ st_w, const float* __restrict__ st_b,
                             __bf16* __restrict__ Asp, __bf16* __restrict__ X1) {
  int b = blockIdx.x;
  const float* row = sp + (size_t)b * 1297;
  __bf16* arow = Asp + (size_t)b * 1280;
  for (int k = threadIdx.x; k < 1280; k += 256)
    arow[k] = (__bf16)(k < 1276 ? row[16 + k] : 0.f);
  __bf16* xrow = X1 + (size_t)b * 576;
  int t = threadIdx.x;
  if (t < 2) {
    float s = sl_b[t];
    #pragma unroll
    for (int k = 0; k < 4; ++k) s += row[k] * sl_w[t * 4 + k];
    xrow[t] = (__bf16)s;
  } else if (t < 10) {
    int n = t - 2;
    float s = st_b[n];
    #pragma unroll
    for (int k = 0; k < 12; ++k) s += row[4 + k] * st_w[n * 12 + k];
    xrow[2 + n] = (__bf16)s;
  } else if (t < 15) {
    xrow[522 + (t - 10)] = (__bf16)row[1292 + (t - 10)];
  } else if (t < 64) {
    xrow[527 + (t - 15)] = (__bf16)0.f;   // 527..575
  }
}

// ---------------------------------------------------------------------------
// Generic WMMA GEMM:  D[m][Dcol+n] = bf16( sum_k A[m][k]*W[n][k] + bias[n] )
// Block = 8 waves; each wave computes a 32(M) x 64(N) strip: 8 accumulator
// tiles, B fragments reused across 2 A row-subtiles. KS is compile-time:
// the fully unrolled ping-pong double buffer needs no register copies and
// all load offsets become instruction immediates.
// Grid = (M/256, N/64). M % 256 == 0, N % 64 == 0, KS % 32 == 0, KS >= 64.
// ---------------------------------------------------------------------------
template <int KS>
__global__ void __launch_bounds__(256)
gemm_bf16_wmma(const __bf16* __restrict__ A, const __bf16* __restrict__ W,
               const float* __restrict__ bias, __bf16* __restrict__ D,
               int Dstride, int Dcol) {
  constexpr int KITERS = KS / 32;
  const int lane = threadIdx.x & 31;
  const int wave = threadIdx.x >> 5;
  const int g  = lane >> 4;      // half-wave select
  const int ln = lane & 15;
  const int m0 = blockIdx.x * 256 + wave * 32;
  const int n0 = blockIdx.y * 64;

  const __bf16* Ar0 = A + (size_t)(m0 + ln) * KS + 8 * g;        // +k, +k+16
  const __bf16* Ar1 = A + (size_t)(m0 + 16 + ln) * KS + 8 * g;
  const __bf16* Wr0 = W + (size_t)(n0 + ln) * KS + 8 * g;        // j -> +16*KS
  constexpr size_t Wj = (size_t)16 * KS;

  v8f acc0[4] = {v8f{}, v8f{}, v8f{}, v8f{}};
  v8f acc1[4] = {v8f{}, v8f{}, v8f{}, v8f{}};

  v16bf a0[2], a1[2], b[2][4];
  a0[0] = ld_bf16_8x2(Ar0, Ar0 + 16);
  a1[0] = ld_bf16_8x2(Ar1, Ar1 + 16);
  #pragma unroll
  for (int j = 0; j < 4; ++j) b[0][j] = ld_bf16_8x2(Wr0 + (size_t)j * Wj, Wr0 + (size_t)j * Wj + 16);

  #pragma unroll
  for (int kk = 0; kk < KITERS - 1; ++kk) {
    const int cur = kk & 1, nxt = cur ^ 1;
    const int k = (kk + 1) * 32;
    // prefetch next fragments (in flight during WMMAs below)
    a0[nxt] = ld_bf16_8x2(Ar0 + k, Ar0 + k + 16);
    a1[nxt] = ld_bf16_8x2(Ar1 + k, Ar1 + k + 16);
    #pragma unroll
    for (int j = 0; j < 4; ++j) b[nxt][j] = ld_bf16_8x2(Wr0 + (size_t)j * Wj + k, Wr0 + (size_t)j * Wj + k + 16);
    #pragma unroll
    for (int j = 0; j < 4; ++j) {
      acc0[j] = wmma_bf16(a0[cur], b[cur][j], acc0[j]);
      acc1[j] = wmma_bf16(a1[cur], b[cur][j], acc1[j]);
    }
  }
  {
    constexpr int last = (KITERS - 1) & 1;
    #pragma unroll
    for (int j = 0; j < 4; ++j) {
      acc0[j] = wmma_bf16(a0[last], b[last][j], acc0[j]);
      acc1[j] = wmma_bf16(a1[last], b[last][j], acc1[j]);
    }
  }

  #pragma unroll
  for (int j = 0; j < 4; ++j) {
    int n = n0 + 16 * j + ln;
    float bs = bias[n];
    #pragma unroll
    for (int v = 0; v < 8; ++v) {
      int m = m0 + v + 8 * g;               // D layout: VGPR v -> row v + 8*hi
      D[(size_t)m * Dstride + Dcol + n]        = (__bf16)(acc0[j][v] + bs);
      D[(size_t)(m + 16) * Dstride + Dcol + n] = (__bf16)(acc1[j][v] + bs);
    }
  }
}

// ---------------------------------------------------------------------------
// Persistent-tile LSTM. One workgroup owns BM=64 batch rows for all 96 steps.
// LDS: Xb (2 x 64x64 bf16, double-buffered x_t), Hb (2 x 64x256 bf16), Cc
// (64x256 f32). Each wave owns a fixed (16-row tile, 128-hidden-col block):
// per step it hoists its 10 A-fragments ([x_t|h], K=320) into registers once,
// then streams statically ping-ponged weight fragments through 8 WMMA tasks,
// applies activations in-register and writes h_next back to LDS.
// One workgroup barrier per step.
// ---------------------------------------------------------------------------
#define LSTM_BM 64
#define LSTM_LDS_BYTES (2*LSTM_BM*64*2 + 2*LSTM_BM*256*2 + LSTM_BM*256*4)

__global__ void __launch_bounds__(256)
lstm_wmma(const float* __restrict__ temporal,   // [B][T][56] f32
          const __bf16* __restrict__ Wcat,      // [1024][320] bf16
          const float* __restrict__ bcat,       // [1024]
          __bf16* __restrict__ Hout,            // X2 [B][384], h_T -> cols 128..383
          int T) {
  extern __shared__ char smem[];
  __bf16* Xb = (__bf16*)smem;                                  // 2 * 64*64
  __bf16* Hb = (__bf16*)(smem + 2 * LSTM_BM * 64 * 2);         // 2 * 64*256
  float*  Cc = (float*)(smem + 2 * LSTM_BM * 64 * 2 + 2 * LSTM_BM * 256 * 2);

  const int tid  = threadIdx.x;
  const int lane = tid & 31;
  const int wave = tid >> 5;
  const int g  = lane >> 4;
  const int ln = lane & 15;
  const int b0 = blockIdx.x * LSTM_BM;

  for (int i = tid; i < LSTM_BM * 256; i += 256) { Hb[i] = (__bf16)0.f; Cc[i] = 0.f; }
  for (int i = tid; i < 2 * LSTM_BM * 64; i += 256) Xb[i] = (__bf16)0.f;
  __syncthreads();

  auto load_x = [&](int t, int buf) {
    for (int i = tid; i < LSTM_BM * 56; i += 256) {
      int m = i / 56, c = i - m * 56;
      Xb[buf * LSTM_BM * 64 + m * 64 + c] =
          (__bf16)temporal[(size_t)(b0 + m) * T * 56 + (size_t)t * 56 + c];
    }
  };
  load_x(0, 0);
  __syncthreads();

  const int mt   = wave >> 1;          // row tile 0..3
  const int ngb  = (wave & 1) * 8;     // hidden block base
  const int mrow = mt * 16 + ln;

  for (int t = 0; t < T; ++t) {
    const int cur = t & 1, nxt = cur ^ 1;
    if (t + 1 < T) load_x(t + 1, nxt);           // overlap next x_t staging
    const __bf16* Xc = Xb + cur * LSTM_BM * 64;
    const __bf16* Hc = Hb + cur * LSTM_BM * 256;
    __bf16*       Hn = Hb + nxt * LSTM_BM * 256;

    // Hoist this wave's A fragments for the whole step (independent of n).
    v16bf afr[10];
    #pragma unroll
    for (int kk = 0; kk < 10; ++kk) {
      const int k = kk * 32;
      const __bf16* arow;
      int o0, o1;
      if (k < 64) { arow = Xc + mrow * 64;  o0 = k + 8 * g;        o1 = k + 16 + 8 * g; }
      else        { arow = Hc + mrow * 256; o0 = (k - 64) + 8 * g; o1 = (k - 64) + 16 + 8 * g; }
      afr[kk] = ld_bf16_8x2(arow + o0, arow + o1);
    }

    for (int ngi = 0; ngi < 8; ++ngi) {
      const int n = (ngb + ngi) * 16;            // hidden column base
      const __bf16* wbase = Wcat + (size_t)(n + ln) * 320 + 8 * g;
      v8f acc[4] = {v8f{}, v8f{}, v8f{}, v8f{}}; // i, f, g, o quadrants

      v16bf bb[2][4];
      #pragma unroll
      for (int q = 0; q < 4; ++q) bb[0][q] = ld_bf16_8x2(wbase + (size_t)q * 256 * 320, wbase + (size_t)q * 256 * 320 + 16);

      #pragma unroll
      for (int kk = 0; kk < 10; ++kk) {
        const int c2 = kk & 1, n2 = c2 ^ 1;
        if (kk < 9) {
          #pragma unroll
          for (int q = 0; q < 4; ++q)
            bb[n2][q] = ld_bf16_8x2(wbase + (size_t)q * 256 * 320 + (kk + 1) * 32, wbase + (size_t)q * 256 * 320 + (kk + 1) * 32 + 16);
        }
        #pragma unroll
        for (int q = 0; q < 4; ++q) acc[q] = wmma_bf16(afr[kk], bb[c2][q], acc[q]);
      }

      const int col = n + ln;
      const float bi = bcat[col], bff = bcat[256 + col];
      const float bg = bcat[512 + col], bo = bcat[768 + col];
      #pragma unroll
      for (int v = 0; v < 8; ++v) {
        const int m = mt * 16 + v + 8 * g;
        const float si = sigm(acc[0][v] + bi);
        const float sf = sigm(acc[1][v] + bff);
        const float tg = fast_tanh(acc[2][v] + bg);
        const float so = sigm(acc[3][v] + bo);
        const float c  = sf * Cc[m * 256 + col] + si * tg;
        Cc[m * 256 + col] = c;
        Hn[m * 256 + col] = (__bf16)(so * fast_tanh(c));
      }
    }
    __syncthreads();
  }

  const __bf16* Hf = Hb + (T & 1) * LSTM_BM * 256;
  for (int i = tid; i < LSTM_BM * 256; i += 256) {
    int m = i >> 8, c = i & 255;
    Hout[(size_t)(b0 + m) * 384 + 128 + c] = Hf[i];
  }
}

// Final l7: out[b] = X5[b][:] . l7_w + l7_b  (K=128), vector loads
__global__ void head_l7(const __bf16* __restrict__ X5, const float* __restrict__ w,
                        const float* __restrict__ bptr, float* __restrict__ out) {
  int b = blockIdx.x * 256 + threadIdx.x;
  if (b >= BDIM) return;
  const v8bf* xr = reinterpret_cast<const v8bf*>(X5 + (size_t)b * 128);
  float s = bptr[0];
  #pragma unroll
  for (int c = 0; c < 16; ++c) {
    v8bf x = xr[c];
    #pragma unroll
    for (int e = 0; e < 8; ++e) s += (float)x[e] * w[c * 8 + e];
  }
  out[b] = s;
}

// ---------------------------------------------------------------------------
// Host launcher
// ---------------------------------------------------------------------------
static char* carve(void* ws, size_t& off, size_t bytes) {
  off = (off + 255) & ~(size_t)255;
  char* p = (char*)ws + off;
  off += bytes;
  return p;
}

extern "C" void kernel_launch(void* const* d_in, const int* /*in_sizes*/, int /*n_in*/,
                              void* d_out, int /*out_size*/, void* d_ws, size_t /*ws_size*/,
                              hipStream_t stream) {
  const float* spatial  = (const float*)d_in[0];
  const float* temporal = (const float*)d_in[1];
  const float* sl_w = (const float*)d_in[2];  const float* sl_b = (const float*)d_in[3];
  const float* st_w = (const float*)d_in[4];  const float* st_b = (const float*)d_in[5];
  const float* seg_w = (const float*)d_in[6]; const float* seg_b = (const float*)d_in[7];
  const float* l1_w = (const float*)d_in[8];  const float* l1_b = (const float*)d_in[9];
  const float* l2_w = (const float*)d_in[10]; const float* l2_b = (const float*)d_in[11];
  const float* w_ih = (const float*)d_in[12]; const float* b_ih = (const float*)d_in[13];
  const float* w_hh = (const float*)d_in[14]; const float* b_hh = (const float*)d_in[15];
  const float* l4_w = (const float*)d_in[16]; const float* l4_b = (const float*)d_in[17];
  const float* l5_w = (const float*)d_in[18]; const float* l5_b = (const float*)d_in[19];
  const float* l6_w = (const float*)d_in[20]; const float* l6_b = (const float*)d_in[21];
  const float* l7_w = (const float*)d_in[22]; const float* l7_b = (const float*)d_in[23];
  float* out = (float*)d_out;

  size_t off = 0;
  __bf16* Wcat = (__bf16*)carve(d_ws, off, (size_t)1024 * 320 * 2);
  float*  bcat = (float*) carve(d_ws, off, 1024 * 4);
  __bf16* segp = (__bf16*)carve(d_ws, off, (size_t)512 * 1280 * 2);
  __bf16* l1p  = (__bf16*)carve(d_ws, off, (size_t)256 * 576 * 2);
  __bf16* l2p  = (__bf16*)carve(d_ws, off, (size_t)128 * 256 * 2);
  __bf16* l4p  = (__bf16*)carve(d_ws, off, (size_t)256 * 384 * 2);
  __bf16* l5p  = (__bf16*)carve(d_ws, off, (size_t)128 * 256 * 2);
  __bf16* l6p  = (__bf16*)carve(d_ws, off, (size_t)128 * 128 * 2);
  __bf16* Asp  = (__bf16*)carve(d_ws, off, (size_t)BDIM * 1280 * 2);  // seg GEMM input
  __bf16* X1   = (__bf16*)carve(d_ws, off, (size_t)BDIM * 576 * 2);   // concat -> l1
  __bf16* O4   = (__bf16*)carve(d_ws, off, (size_t)BDIM * 256 * 2);   // l1 out
  __bf16* X2   = (__bf16*)carve(d_ws, off, (size_t)BDIM * 384 * 2);   // [out5 | h_T]
  // Asp is dead after the seg GEMM -> reuse for head activations
  __bf16* X3 = Asp;                                            // l4 out [B][256]
  __bf16* X4 = Asp + (size_t)BDIM * 256;                       // l5 out [B][128]
  __bf16* X5 = Asp + (size_t)BDIM * 256 + (size_t)BDIM * 128;  // l6 out [B][128]

  auto pw = [&](const float* W, __bf16* Wp, int N, int K, int Kp) {
    int total = N * Kp;
    pack_weight<<<(total + 255) / 256, 256, 0, stream>>>(W, Wp, N, K, Kp);
  };
  pw(seg_w, segp, 512, 1276, 1280);
  pw(l1_w,  l1p,  256, 527,  576);
  pw(l2_w,  l2p,  128, 256,  256);
  pw(l4_w,  l4p,  256, 384,  384);
  pw(l5_w,  l5p,  128, 256,  256);
  pw(l6_w,  l6p,  128, 128,  128);
  pack_lstm<<<(1024 * 320 + 255) / 256, 256, 0, stream>>>(w_ih, w_hh, b_ih, b_hh, Wcat, bcat);
  pack_spatial<<<BDIM, 256, 0, stream>>>(spatial, sl_w, sl_b, st_w, st_b, Asp, X1);

  // spatial chain: seg -> l1 -> l2 (out5 into X2 cols 0..127)
  gemm_bf16_wmma<1280><<<dim3(BDIM / 256, 512 / 64), 256, 0, stream>>>(Asp, segp, seg_b, X1, 576, 10);
  gemm_bf16_wmma<576><<<dim3(BDIM / 256, 256 / 64), 256, 0, stream>>>(X1, l1p, l1_b, O4, 256, 0);
  gemm_bf16_wmma<256><<<dim3(BDIM / 256, 128 / 64), 256, 0, stream>>>(O4, l2p, l2_b, X2, 384, 0);

  // LSTM: h_T into X2 cols 128..383
  lstm_wmma<<<BDIM / LSTM_BM, 256, LSTM_LDS_BYTES, stream>>>(temporal, Wcat, bcat, X2, TSTEPS);

  // head: l4 -> l5 -> l6 -> l7
  gemm_bf16_wmma<384><<<dim3(BDIM / 256, 256 / 64), 256, 0, stream>>>(X2, l4p, l4_b, X3, 256, 0);
  gemm_bf16_wmma<256><<<dim3(BDIM / 256, 128 / 64), 256, 0, stream>>>(X3, l5p, l5_b, X4, 128, 0);
  gemm_bf16_wmma<128><<<dim3(BDIM / 256, 128 / 64), 256, 0, stream>>>(X4, l6p, l6_b, X5, 128, 0);
  head_l7<<<(BDIM + 255) / 256, 256, 0, stream>>>(X5, l7_w, l7_b, out);
}
